// LinearSelfAttention_5540507811845
// MI455X (gfx1250) — matrix operand order repaired and checked by reading, not compiled, not run on hardware
//
#include <hip/hip_runtime.h>
#include <cstdint>
#include <cstddef>

typedef __bf16 bf16_t;
typedef __attribute__((ext_vector_type(8)))  __bf16 v8bf;
typedef __attribute__((ext_vector_type(16))) __bf16 v16bf;
typedef __attribute__((ext_vector_type(8)))  float  v8f;

#define NB 4
#define NL 8192
#define ND 512
#define NH 8
#define DH 64
#define M1 (NB*NL)   /* 32768 rows of x */
#define N1 (3*ND)    /* 1536 qkv cols  */

__device__ __forceinline__ v16bf cat16(v8bf lo, v8bf hi) {
  return __builtin_shufflevector(lo, hi, 0,1,2,3,4,5,6,7,8,9,10,11,12,13,14,15);
}
__device__ __forceinline__ v8f wmma_bf16(v16bf a, v16bf b, v8f c) {
  // D = A(16x32) * B(32x16) + C, f32 accumulate
  return __builtin_amdgcn_wmma_f32_16x16x32_bf16(false, a, false, b, (short)0, c, false, false);
}

// Async copy of 16 bytes global -> LDS (CDNA5 GLOBAL_LOAD_ASYNC_TO_LDS_B128).
// LDS generic pointers carry the LDS byte offset in addr[31:0] (aperture rule),
// so truncation yields the DS address the instruction wants in VDST.
__device__ __forceinline__ void async_copy_b128(const void* gsrc, void* ldst) {
  unsigned lds_addr = (unsigned)(unsigned long long)ldst;
  asm volatile("global_load_async_to_lds_b128 %0, %1, off"
               :: "v"(lds_addr), "v"(gsrc) : "memory");
}
__device__ __forceinline__ void wait_async(int n) {
  if (n == 0) asm volatile("s_wait_asynccnt 0x0" ::: "memory");
  else        asm volatile("s_wait_asynccnt 0x4" ::: "memory");
}

// ---------------------------------------------------------------------------
// Weight transpose + fp32 -> bf16 convert: w (K x N row-major) -> wT (N x K)
// ---------------------------------------------------------------------------
__global__ __launch_bounds__(256) void wt_convert_kernel(
    const float* __restrict__ w, bf16_t* __restrict__ wT, int K, int N)
{
  int idx = blockIdx.x * 256 + threadIdx.x;
  if (idx >= K * N) return;
  int n = idx % N;
  int kk = idx / N;
  wT[(size_t)n * K + kk] = (bf16_t)w[idx];
}

// ---------------------------------------------------------------------------
// QKV GEMM: (32768x512 fp32 x) @ (512x1536 bf16 wT) + bias, fused phi(elu+1)
// 128x128 block tile, 8 waves each 64x32, BK=32, bf16 WMMA f32 acc.
// A tile: fp32->bf16 convert through VGPRs; B tile: async global->LDS.
// ---------------------------------------------------------------------------
__global__ __launch_bounds__(256) void qkv_gemm_kernel(
    const float* __restrict__ x, const bf16_t* __restrict__ wT,
    const float* __restrict__ bias,
    bf16_t* __restrict__ q, bf16_t* __restrict__ k, bf16_t* __restrict__ v)
{
  __shared__ alignas(16) bf16_t ldsA[128 * 32];
  __shared__ alignas(16) bf16_t ldsB[128 * 32];
  const int tid    = threadIdx.x;
  const int lane   = tid & 31;
  const int wave   = tid >> 5;
  const int mBlock = blockIdx.y * 128;
  const int nBlock = blockIdx.x * 128;
  const int wm   = (wave >> 2) * 64;   // 0 / 64
  const int wn   = (wave & 3) * 32;    // 0,32,64,96
  const int l16  = lane & 15;
  const int hsel = lane >> 4;          // which half of the wave

  const v8f zero8 = {0.f,0.f,0.f,0.f,0.f,0.f,0.f,0.f};
  v8f acc[4][2];
  #pragma unroll
  for (int i = 0; i < 4; ++i)
    #pragma unroll
    for (int j = 0; j < 2; ++j) acc[i][j] = zero8;

  const int sr = tid >> 2;        // staging row 0..63 (+64 on rep 1)
  const int sc = (tid & 3) * 8;   // staging col (bf16 elements)

  for (int kk = 0; kk < ND; kk += 32) {
    // ---- stage B async (bf16 copy), A via VGPR convert ----
    #pragma unroll
    for (int rep = 0; rep < 2; ++rep) {
      const int row = sr + rep * 64;
      async_copy_b128(wT + (size_t)(nBlock + row) * ND + kk + sc,
                      &ldsB[row * 32 + sc]);
      const float* ap = x + (size_t)(mBlock + row) * ND + kk + sc;
      float4 f0 = *(const float4*)ap;
      float4 f1 = *(const float4*)(ap + 4);
      bf16_t* dst = &ldsA[row * 32 + sc];
      dst[0]=(bf16_t)f0.x; dst[1]=(bf16_t)f0.y; dst[2]=(bf16_t)f0.z; dst[3]=(bf16_t)f0.w;
      dst[4]=(bf16_t)f1.x; dst[5]=(bf16_t)f1.y; dst[6]=(bf16_t)f1.z; dst[7]=(bf16_t)f1.w;
    }
    if (kk + 32 < ND) {
      __builtin_prefetch(x + (size_t)(mBlock + sr) * ND + kk + 32 + sc, 0, 1);
    }
    wait_async(0);
    __syncthreads();

    // ---- fragments per ISA layout ----
    v16bf afr[4];
    #pragma unroll
    for (int i = 0; i < 4; ++i) {
      const bf16_t* ap = &ldsA[(wm + i * 16 + l16) * 32 + hsel * 8];
      afr[i] = cat16(*(const v8bf*)ap, *(const v8bf*)(ap + 16));
    }
    v16bf bfr[2];
    #pragma unroll
    for (int j = 0; j < 2; ++j) {
      const bf16_t* bp = &ldsB[(wn + j * 16 + l16) * 32 + hsel * 8];
      bfr[j] = cat16(*(const v8bf*)bp, *(const v8bf*)(bp + 16));
    }
    #pragma unroll
    for (int i = 0; i < 4; ++i)
      #pragma unroll
      for (int j = 0; j < 2; ++j)
        acc[i][j] = wmma_bf16(afr[i], bfr[j], acc[i][j]);
    __syncthreads();
  }

  // ---- epilogue: bias + phi for q,k; plain bias for v; bf16 stores ----
  #pragma unroll
  for (int i = 0; i < 4; ++i) {
    #pragma unroll
    for (int j = 0; j < 2; ++j) {
      const int col = nBlock + wn + j * 16 + l16;
      const float bval = bias[col];
      #pragma unroll
      for (int r = 0; r < 8; ++r) {
        const int row = mBlock + wm + i * 16 + r + hsel * 8;
        const float y = acc[i][j][r] + bval;
        const size_t rb = (size_t)row * ND;
        if (col < ND) {
          const float p = (y > 0.f) ? (y + 1.f) : __expf(y);
          q[rb + col] = (bf16_t)p;
        } else if (col < 2 * ND) {
          const float p = (y > 0.f) ? (y + 1.f) : __expf(y);
          k[rb + (col - ND)] = (bf16_t)p;
        } else {
          v[rb + (col - 2 * ND)] = (bf16_t)y;
        }
      }
    }
  }
}

// ---------------------------------------------------------------------------
// kv[bh, m, d] = sum_l k[l,d] * v[l,m].  Block = (bh, group of 8 m), 64 thr(d).
// ---------------------------------------------------------------------------
__global__ __launch_bounds__(64) void kv_kernel(
    const bf16_t* __restrict__ k, const bf16_t* __restrict__ v,
    bf16_t* __restrict__ kv)
{
  const int mg = blockIdx.x & 7;
  const int bh = blockIdx.x >> 3;
  const int b = bh >> 3, h = bh & 7;
  const int m0 = mg * 8;
  const int d = threadIdx.x;
  const bf16_t* kbase = k + (size_t)b * NL * ND + h * DH;
  const bf16_t* vbase = v + (size_t)b * NL * ND + h * DH;
  __shared__ float vs[64][8];
  float acc[8];
  #pragma unroll
  for (int mm = 0; mm < 8; ++mm) acc[mm] = 0.f;

  for (int l0 = 0; l0 < NL; l0 += 64) {
    __syncthreads();
    v8bf vv = *(const v8bf*)(vbase + (size_t)(l0 + d) * ND + m0);
    #pragma unroll
    for (int mm = 0; mm < 8; ++mm) vs[d][mm] = (float)vv[mm];
    __syncthreads();
    for (int j = 0; j < 64; ++j) {
      const float kd = (float)kbase[(size_t)(l0 + j) * ND + d];
      #pragma unroll
      for (int mm = 0; mm < 8; ++mm) acc[mm] += kd * vs[j][mm];
    }
  }
  #pragma unroll
  for (int mm = 0; mm < 8; ++mm)
    kv[(size_t)bh * DH * DH + (size_t)(m0 + mm) * DH + d] = (bf16_t)acc[mm];
}

// ---------------------------------------------------------------------------
// ksum_eps[bh, d] = sum_l k[l,d] + 1e-6
// ---------------------------------------------------------------------------
__global__ __launch_bounds__(64) void ksum_kernel(
    const bf16_t* __restrict__ k, float* __restrict__ ksum_eps)
{
  const int bh = blockIdx.x;
  const int b = bh >> 3, h = bh & 7;
  const int d = threadIdx.x;
  const bf16_t* kbase = k + (size_t)b * NL * ND + h * DH;
  float s = 0.f;
  for (int l = 0; l < NL; ++l) s += (float)kbase[(size_t)l * ND + d];
  ksum_eps[bh * DH + d] = s + 1e-6f;
}

// ---------------------------------------------------------------------------
// denom[bh, l] = sum_d q[l,d] * ksum_eps[bh,d]
// ---------------------------------------------------------------------------
__global__ __launch_bounds__(256) void denom_kernel(
    const bf16_t* __restrict__ q, const float* __restrict__ ksum_eps,
    float* __restrict__ denom)
{
  const int idx = blockIdx.x * 256 + threadIdx.x;   // bh*NL + l
  const int bh = idx >> 13;
  const int l  = idx & (NL - 1);
  const int b = bh >> 3, h = bh & 7;
  const bf16_t* qrow = q + ((size_t)(b * NL + l)) * ND + h * DH;
  const float* ks = ksum_eps + bh * DH;
  float s = 0.f;
  #pragma unroll
  for (int d8 = 0; d8 < 8; ++d8) {
    v8bf qq = *(const v8bf*)(qrow + d8 * 8);
    #pragma unroll
    for (int e = 0; e < 8; ++e) s += (float)qq[e] * ks[d8 * 8 + e];
  }
  denom[idx] = s;
}

// ---------------------------------------------------------------------------
// attn[l, h*64+m] = (sum_d q[l,d]*kv[m,d]) / denom[l].  WMMA, A/B direct
// from global (layouts already match fragment needs). Block: (bh, 128 rows).
// ---------------------------------------------------------------------------
__global__ __launch_bounds__(256) void attn_kernel(
    const bf16_t* __restrict__ q, const bf16_t* __restrict__ kv,
    const float* __restrict__ denom, bf16_t* __restrict__ attn)
{
  const int blk = blockIdx.x;
  const int mt = blk & (NL / 128 - 1);
  const int bh = blk >> 6;
  const int b = bh >> 3, h = bh & 7;
  const int lane = threadIdx.x & 31;
  const int wave = threadIdx.x >> 5;
  const int l16 = lane & 15, hsel = lane >> 4;
  const int l0 = mt * 128 + wave * 16;

  const bf16_t* qp  = q  + ((size_t)(b * NL + l0 + l16)) * ND + h * DH;
  const bf16_t* kvb = kv + (size_t)bh * DH * DH;

  const v8f zero8 = {0.f,0.f,0.f,0.f,0.f,0.f,0.f,0.f};
  v8f acc[4];
  #pragma unroll
  for (int j = 0; j < 4; ++j) acc[j] = zero8;

  #pragma unroll
  for (int kk = 0; kk < DH; kk += 32) {
    const bf16_t* ap = qp + kk + hsel * 8;
    v16bf a = cat16(*(const v8bf*)ap, *(const v8bf*)(ap + 16));
    #pragma unroll
    for (int j = 0; j < 4; ++j) {
      const bf16_t* bp = kvb + (size_t)(j * 16 + l16) * DH + kk + hsel * 8;
      v16bf bfr = cat16(*(const v8bf*)bp, *(const v8bf*)(bp + 16));
      acc[j] = wmma_bf16(a, bfr, acc[j]);
    }
  }

  #pragma unroll
  for (int j = 0; j < 4; ++j) {
    const int m = j * 16 + l16;
    #pragma unroll
    for (int r = 0; r < 8; ++r) {
      const int l = l0 + r + hsel * 8;
      const float dn = denom[(size_t)bh * NL + l];
      const float val = acc[j][r] / dn;
      attn[((size_t)(b * NL + l)) * ND + h * DH + m] = (bf16_t)val;
    }
  }
}

// ---------------------------------------------------------------------------
// Output GEMM: (32768x512 bf16 attn) @ (512x512 bf16 woT) + b_out -> fp32 out
// Fully async, double-buffered LDS staging: tile t+1 in flight during tile t
// WMMA work (4 async b128 per thread per tile; s_wait_asynccnt 4 pipelines).
// ---------------------------------------------------------------------------
__global__ __launch_bounds__(256) void out_gemm_kernel(
    const bf16_t* __restrict__ attn, const bf16_t* __restrict__ wT,
    const float* __restrict__ bias, float* __restrict__ out)
{
  __shared__ alignas(16) bf16_t ldsA[2][128 * 32];
  __shared__ alignas(16) bf16_t ldsB[2][128 * 32];
  const int tid    = threadIdx.x;
  const int lane   = tid & 31;
  const int wave   = tid >> 5;
  const int mBlock = blockIdx.y * 128;
  const int nBlock = blockIdx.x * 128;
  const int wm   = (wave >> 2) * 64;
  const int wn   = (wave & 3) * 32;
  const int l16  = lane & 15;
  const int hsel = lane >> 4;

  const v8f zero8 = {0.f,0.f,0.f,0.f,0.f,0.f,0.f,0.f};
  v8f acc[4][2];
  #pragma unroll
  for (int i = 0; i < 4; ++i)
    #pragma unroll
    for (int j = 0; j < 2; ++j) acc[i][j] = zero8;

  const int sr = tid >> 2;
  const int sc = (tid & 3) * 8;

  // issue async stage of one 128x32 A tile + 128x32 B tile into buffer `buf`
  auto stage = [&](int buf, int kk) {
    #pragma unroll
    for (int rep = 0; rep < 2; ++rep) {
      const int row = sr + rep * 64;
      async_copy_b128(attn + (size_t)(mBlock + row) * ND + kk + sc,
                      &ldsA[buf][row * 32 + sc]);
      async_copy_b128(wT + (size_t)(nBlock + row) * ND + kk + sc,
                      &ldsB[buf][row * 32 + sc]);
    }
  };

  stage(0, 0);
  for (int kk = 0, t = 0; kk < ND; kk += 32, ++t) {
    const int cur = t & 1;
    const bool more = (kk + 32 < ND);
    if (more) stage(cur ^ 1, kk + 32);
    wait_async(more ? 4 : 0);   // current tile's 4 asyncs done; next 4 in flight
    __syncthreads();

    v16bf afr[4];
    #pragma unroll
    for (int i = 0; i < 4; ++i) {
      const bf16_t* ap = &ldsA[cur][(wm + i * 16 + l16) * 32 + hsel * 8];
      afr[i] = cat16(*(const v8bf*)ap, *(const v8bf*)(ap + 16));
    }
    v16bf bfr[2];
    #pragma unroll
    for (int j = 0; j < 2; ++j) {
      const bf16_t* bp = &ldsB[cur][(wn + j * 16 + l16) * 32 + hsel * 8];
      bfr[j] = cat16(*(const v8bf*)bp, *(const v8bf*)(bp + 16));
    }
    #pragma unroll
    for (int i = 0; i < 4; ++i)
      #pragma unroll
      for (int j = 0; j < 2; ++j)
        acc[i][j] = wmma_bf16(afr[i], bfr[j], acc[i][j]);
    __syncthreads();   // all waves done reading buf `cur` before t+2 overwrites
  }

  #pragma unroll
  for (int i = 0; i < 4; ++i) {
    #pragma unroll
    for (int j = 0; j < 2; ++j) {
      const int col = nBlock + wn + j * 16 + l16;
      const float bval = bias[col];
      #pragma unroll
      for (int r = 0; r < 8; ++r) {
        const int row = mBlock + wm + i * 16 + r + hsel * 8;
        out[(size_t)row * ND + col] = acc[i][j][r] + bval;
      }
    }
  }
}

// ---------------------------------------------------------------------------
extern "C" void kernel_launch(void* const* d_in, const int* in_sizes, int n_in,
                              void* d_out, int out_size, void* d_ws, size_t ws_size,
                              hipStream_t stream) {
  (void)in_sizes; (void)n_in; (void)out_size; (void)ws_size;
  const float* x     = (const float*)d_in[0];
  const float* w_qkv = (const float*)d_in[1];
  const float* b_qkv = (const float*)d_in[2];
  const float* w_out = (const float*)d_in[3];
  const float* b_out = (const float*)d_in[4];
  float* out = (float*)d_out;

  char* ws = (char*)d_ws;
  size_t off = 0;
  auto take = [&](size_t bytes) -> void* {
    void* p = (void*)(ws + off);
    off += (bytes + 255) & ~((size_t)255);
    return p;
  };
  bf16_t* qb     = (bf16_t*)take((size_t)M1 * ND * sizeof(bf16_t));
  bf16_t* kb     = (bf16_t*)take((size_t)M1 * ND * sizeof(bf16_t));
  bf16_t* vb     = (bf16_t*)take((size_t)M1 * ND * sizeof(bf16_t));
  bf16_t* attnb  = (bf16_t*)take((size_t)M1 * ND * sizeof(bf16_t));
  bf16_t* wqkvT  = (bf16_t*)take((size_t)ND * N1 * sizeof(bf16_t));
  bf16_t* woutT  = (bf16_t*)take((size_t)ND * ND * sizeof(bf16_t));
  bf16_t* kvb    = (bf16_t*)take((size_t)NB * NH * DH * DH * sizeof(bf16_t));
  float*  ksum   = (float*)take((size_t)NB * NH * DH * sizeof(float));
  float*  denomb = (float*)take((size_t)NB * NH * NL * sizeof(float));

  // 1) weight transpose + bf16 convert
  wt_convert_kernel<<<(ND * N1 + 255) / 256, 256, 0, stream>>>(w_qkv, wqkvT, ND, N1);
  wt_convert_kernel<<<(ND * ND + 255) / 256, 256, 0, stream>>>(w_out, woutT, ND, ND);
  // 2) QKV GEMM + bias + phi
  qkv_gemm_kernel<<<dim3(N1 / 128, M1 / 128), 256, 0, stream>>>(
      x, wqkvT, b_qkv, qb, kb, vb);
  // 3) kv outer-product accumulation
  kv_kernel<<<NB * NH * 8, 64, 0, stream>>>(kb, vb, kvb);
  // 4) k_sum + eps
  ksum_kernel<<<NB * NH, 64, 0, stream>>>(kb, ksum);
  // 5) denominator
  denom_kernel<<<(NB * NH * NL) / 256, 256, 0, stream>>>(qb, ksum, denomb);
  // 6) attn = (Q @ kv^T) / denom
  attn_kernel<<<NB * NH * (NL / 128), 256, 0, stream>>>(qb, kvb, denomb, attnb);
  // 7) output GEMM + bias
  out_gemm_kernel<<<dim3(ND / 128, M1 / 128), 256, 0, stream>>>(
      attnb, woutT, b_out, out);
}
